// HistoryAttention_56762287784312
// MI455X (gfx1250) — matrix operand order repaired and checked by reading, not compiled, not run on hardware
//
#include <hip/hip_runtime.h>

typedef float v4f __attribute__((ext_vector_type(4)));
typedef float v8f __attribute__((ext_vector_type(8)));
typedef __bf16 v4bf __attribute__((ext_vector_type(4)));
typedef __bf16 v8bf __attribute__((ext_vector_type(8)));
typedef __bf16 v16bf __attribute__((ext_vector_type(16)));

#define D_DIM 1024
#define B_SZ 4
#define T_SZ 2048
#define H_SZ 32
#define TT 4                     // t-positions per attention block
#define ROWS 128                 // TT*H_SZ rows per block GEMM
#define NC 64                    // output-column chunk
#define KE 32                    // K-panel depth (== WMMA bf16 K)
#define NTHREADS 256

// Per-buffer panel layout (bytes, relative to buffer base)
#define AS_H_OFF 0
#define AS_L_OFF (ROWS * KE * 2)                    //  8192
#define WS_H_OFF (2 * ROWS * KE * 2)                // 16384
#define WS_L_OFF (2 * ROWS * KE * 2 + NC * KE * 2)  // 20480
#define PANEL_BYTES (2 * ROWS * KE * 2 + 2 * NC * KE * 2)  // 24576
#define GEMM_SMEM (2 * PANEL_BYTES)                 // 49152 (double buffer)
#define CC_BYTES (ROWS * NC * 4)                    // 32768 (aliases panels)

// ---------------------------------------------------------------------------
// Software-pipelined staging: global->VGPR (issued ahead of the WMMAs), then
// bf16 hi/lo split + ds_store into the spare panel buffer.
// ---------------------------------------------------------------------------
struct StageRegs { v4f a[4]; v4f w[2]; };

__device__ __forceinline__ StageRegs load_stage(const float* __restrict__ A,
                                                const float* __restrict__ W,
                                                int e0) {
  const int tid = threadIdx.x;
  StageRegs s;
#pragma unroll
  for (int i = 0; i < 4; ++i) {               // A: 128x32 f32 = 1024 float4
    int f = tid + NTHREADS * i;
    int r = f >> 3, c4 = (f & 7) << 2;
    s.a[i] = *(const v4f*)(A + (size_t)r * D_DIM + e0 + c4);
  }
#pragma unroll
  for (int i = 0; i < 2; ++i) {               // W: 64x32 f32 = 512 float4
    int f = tid + NTHREADS * i;
    int r = f >> 3, c4 = (f & 7) << 2;
    s.w[i] = *(const v4f*)(W + (size_t)r * D_DIM + e0 + c4);
  }
  return s;
}

__device__ __forceinline__ void split4(const v4f v, v4bf* h, v4bf* l) {
#pragma unroll
  for (int j = 0; j < 4; ++j) {
    __bf16 hb = (__bf16)v[j];
    (*h)[j] = hb;
    (*l)[j] = (__bf16)(v[j] - (float)hb);
  }
}

__device__ __forceinline__ void store_stage(const StageRegs& s, char* pbuf) {
  __bf16* AsH = (__bf16*)(pbuf + AS_H_OFF);
  __bf16* AsL = (__bf16*)(pbuf + AS_L_OFF);
  __bf16* WsH = (__bf16*)(pbuf + WS_H_OFF);
  __bf16* WsL = (__bf16*)(pbuf + WS_L_OFF);
  const int tid = threadIdx.x;
#pragma unroll
  for (int i = 0; i < 4; ++i) {
    int f = tid + NTHREADS * i;
    int r = f >> 3, c4 = (f & 7) << 2;
    v4bf h, l;
    split4(s.a[i], &h, &l);
    *(v4bf*)(AsH + r * KE + c4) = h;
    *(v4bf*)(AsL + r * KE + c4) = l;
  }
#pragma unroll
  for (int i = 0; i < 2; ++i) {
    int f = tid + NTHREADS * i;
    int r = f >> 3, c4 = (f & 7) << 2;
    v4bf h, l;
    split4(s.w[i], &h, &l);
    *(v4bf*)(WsH + r * KE + c4) = h;
    *(v4bf*)(WsL + r * KE + c4) = l;
  }
}

// A frag (16x32 bf16): lanes 0-15 hold K=0-7 & 16-23; lanes 16-31 K=8-15 & 24-31
__device__ __forceinline__ v16bf load_a_frag(const __bf16* P, int row, int ah) {
  const __bf16* p = P + row * KE + ah;        // ah = 0 or 8 elems
  v8bf p0 = *(const v8bf*)(p);
  v8bf p1 = *(const v8bf*)(p + 16);
  return __builtin_shufflevector(p0, p1, 0, 1, 2, 3, 4, 5, 6, 7,
                                 8, 9, 10, 11, 12, 13, 14, 15);
}
// B frag (32x16 bf16): lane n; lanes 0-15 hold K=0-7 & 16-23; lanes 16-31 K=8-15 & 24-31
__device__ __forceinline__ v16bf load_b_frag(const __bf16* P, int n, int bh) {
  const __bf16* p = P + n * KE + bh;          // bh = 0 or 8 elems
  v8bf p0 = *(const v8bf*)(p);
  v8bf p1 = *(const v8bf*)(p + 16);
  return __builtin_shufflevector(p0, p1, 0, 1, 2, 3, 4, 5, 6, 7,
                                 8, 9, 10, 11, 12, 13, 14, 15);
}

// Per-wave bf16x3 MMA over one staged panel; wave owns 32x32 (2x2 tiles).
__device__ __forceinline__ void wave_mma_panel(const char* pbuf, int waveRow,
                                               int waveCol, int lane,
                                               v8f c[2][2]) {
  const __bf16* AsH = (const __bf16*)(pbuf + AS_H_OFF);
  const __bf16* AsL = (const __bf16*)(pbuf + AS_L_OFF);
  const __bf16* WsH = (const __bf16*)(pbuf + WS_H_OFF);
  const __bf16* WsL = (const __bf16*)(pbuf + WS_L_OFF);
  const int lm = lane & 15;
  const int ah = (lane >> 4) << 3;
  const int bh = (lane >> 4) << 3;
  v16bf aH[2], aL[2], bH[2], bL[2];
#pragma unroll
  for (int mt = 0; mt < 2; ++mt) {
    aH[mt] = load_a_frag(AsH, waveRow + mt * 16 + lm, ah);
    aL[mt] = load_a_frag(AsL, waveRow + mt * 16 + lm, ah);
  }
#pragma unroll
  for (int nt = 0; nt < 2; ++nt) {
    bH[nt] = load_b_frag(WsH, waveCol + nt * 16 + lm, bh);
    bL[nt] = load_b_frag(WsL, waveCol + nt * 16 + lm, bh);
  }
#pragma unroll
  for (int mt = 0; mt < 2; ++mt)
#pragma unroll
    for (int nt = 0; nt < 2; ++nt) {
      c[mt][nt] = __builtin_amdgcn_wmma_f32_16x16x32_bf16(
          false, aH[mt], false, bH[nt], (short)0, c[mt][nt], false, false);
      c[mt][nt] = __builtin_amdgcn_wmma_f32_16x16x32_bf16(
          false, aH[mt], false, bL[nt], (short)0, c[mt][nt], false, false);
      c[mt][nt] = __builtin_amdgcn_wmma_f32_16x16x32_bf16(
          false, aL[mt], false, bH[nt], (short)0, c[mt][nt], false, false);
    }
}

// C[128x64] = Ablk[128x1024] * Wblk[64x1024]^T, bf16x3, double-buffered LDS.
__device__ __forceinline__ void chunk_gemm(const float* __restrict__ Ablk,
                                           const float* __restrict__ Wblk,
                                           char* smem, v8f c[2][2]) {
  const int lane = threadIdx.x & 31;
  const int wave = threadIdx.x >> 5;
  const int waveRow = (wave >> 1) * 32;
  const int waveCol = (wave & 1) * 32;

  StageRegs s = load_stage(Ablk, Wblk, 0);
  store_stage(s, smem);
  __syncthreads();
  int cur = 0;
#pragma unroll 1
  for (int e0 = 0; e0 < D_DIM - KE; e0 += KE) {
    s = load_stage(Ablk, Wblk, e0 + KE);                 // overlaps WMMAs
    wave_mma_panel(smem + cur * PANEL_BYTES, waveRow, waveCol, lane, c);
    store_stage(s, smem + (cur ^ 1) * PANEL_BYTES);
    __syncthreads();                                      // publish + protect
    cur ^= 1;
  }
  wave_mma_panel(smem + cur * PANEL_BYTES, waveRow, waveCol, lane, c);
  __syncthreads();
}

// ---------------------------------------------------------------------------
// Kernels 1 & 3: C[m][n] = A[m][:] . W[n][:] + bias[n]
// grid = (D/NC, M/ROWS), block = 256
// ---------------------------------------------------------------------------
__global__ __launch_bounds__(NTHREADS)
void proj_gemm_kernel(const float* __restrict__ A, const float* __restrict__ W,
                      const float* __restrict__ bias, float* __restrict__ C) {
  __shared__ __align__(16) char smem[GEMM_SMEM];
  const int n0 = blockIdx.x * NC;
  const int m0 = blockIdx.y * ROWS;
  v8f c[2][2] = {};
  chunk_gemm(A + (size_t)m0 * D_DIM, W + (size_t)n0 * D_DIM, smem, c);

  const int lane = threadIdx.x & 31;
  const int wave = threadIdx.x >> 5;
  const int lm = lane & 15;
  const int mh = (lane >> 4) << 3;
  const int waveRow = (wave >> 1) * 32;
  const int waveCol = (wave & 1) * 32;
#pragma unroll
  for (int mt = 0; mt < 2; ++mt)
#pragma unroll
    for (int nt = 0; nt < 2; ++nt) {
      int n = n0 + waveCol + nt * 16 + lm;
      float bvv = bias[n];
      int mbase = m0 + waveRow + mt * 16 + mh;
#pragma unroll
      for (int r = 0; r < 8; ++r)
        C[(size_t)(mbase + r) * D_DIM + n] = c[mt][nt][r] + bvv;
    }
}

// ---------------------------------------------------------------------------
// Kernel 2: fused K/V projection + decayed softmax attention for TT=4
// positions (128 history rows), all batches. Writes Y = attended + x.
// ---------------------------------------------------------------------------
__global__ __launch_bounds__(NTHREADS)
void history_attn_kernel(const float* __restrict__ x,
                         const float* __restrict__ history,
                         const float* __restrict__ Wk, const float* __restrict__ bk,
                         const float* __restrict__ Wv, const float* __restrict__ bv,
                         const float* __restrict__ decay,
                         const float* __restrict__ Q,
                         float* __restrict__ Y) {
  __shared__ __align__(16) char smem[GEMM_SMEM];    // panels ∪ K/V chunk
  __shared__ float Sc[B_SZ * TT * H_SZ];            // 512 floats
  float* Cc = (float*)smem;
  const int t0 = blockIdx.x * TT;
  const int tid = threadIdx.x;
  const int lane = tid & 31;
  const int wave = tid >> 5;
  const int lm = lane & 15;
  const int mh = (lane >> 4) << 3;
  const int waveRow = (wave >> 1) * 32;
  const int waveCol = (wave & 1) * 32;

#pragma unroll
  for (int i = 0; i < 2; ++i) Sc[tid + NTHREADS * i] = 0.0f;
  __syncthreads();

  const float* histBlk = history + (size_t)t0 * H_SZ * D_DIM;

  // ---- Phase 1: K chunks + score accumulation ----
  for (int dc = 0; dc < D_DIM; dc += NC) {
    v8f c[2][2] = {};
    chunk_gemm(histBlk, Wk + (size_t)dc * D_DIM, smem, c);
#pragma unroll
    for (int mt = 0; mt < 2; ++mt)
#pragma unroll
      for (int nt = 0; nt < 2; ++nt) {
        int n = waveCol + nt * 16 + lm;
        float bkv = bk[dc + n];
        int rbase = waveRow + mt * 16 + mh;
#pragma unroll
        for (int r = 0; r < 8; ++r)
          Cc[(rbase + r) * NC + n] = c[mt][nt][r] + bkv;
      }
    __syncthreads();
    // scores[b,tl,h] += Q[b,t0+tl, dc:dc+64] . Kc[tl*32+h, :]
#pragma unroll
    for (int i = 0; i < 2; ++i) {
      int idx = tid + NTHREADS * i;      // b*128 + tl*32 + h
      int b  = idx >> 7;
      int tl = (idx >> 5) & 3;
      int h  = idx & 31;
      const float* q = Q + ((size_t)(b * T_SZ + t0 + tl) * D_DIM + dc);
      const float* k = Cc + (tl * H_SZ + h) * NC;
      float sdot = 0.f;
#pragma unroll
      for (int j = 0; j < NC; ++j) sdot += q[j] * k[j];
      Sc[idx] += sdot;
    }
    __syncthreads();
  }

  // ---- softmax with flipped log-decay bias (one thread per (b,tl)) ----
  if (tid < B_SZ * TT) {
    float* s = Sc + tid * H_SZ;
    const float scale = 0.03125f;        // 1/sqrt(1024)
    float sb[H_SZ];
    float m = -1e30f;
#pragma unroll
    for (int h = 0; h < H_SZ; ++h) {
      float v = s[h] * scale + __logf(decay[H_SZ - 1 - h] + 1e-10f);
      sb[h] = v;
      m = fmaxf(m, v);
    }
    float sum = 0.f;
#pragma unroll
    for (int h = 0; h < H_SZ; ++h) { sb[h] = __expf(sb[h] - m); sum += sb[h]; }
    float inv = 1.0f / sum;
#pragma unroll
    for (int h = 0; h < H_SZ; ++h) s[h] = sb[h] * inv;
  }
  __syncthreads();

  // ---- Phase 2: V chunks + attended + residual ----
  for (int dc = 0; dc < D_DIM; dc += NC) {
    v8f c[2][2] = {};
    chunk_gemm(histBlk, Wv + (size_t)dc * D_DIM, smem, c);
#pragma unroll
    for (int mt = 0; mt < 2; ++mt)
#pragma unroll
      for (int nt = 0; nt < 2; ++nt) {
        int n = waveCol + nt * 16 + lm;
        float bvv = bv[dc + n];
        int rbase = waveRow + mt * 16 + mh;
#pragma unroll
        for (int r = 0; r < 8; ++r)
          Cc[(rbase + r) * NC + n] = c[mt][nt][r] + bvv;
      }
    __syncthreads();
#pragma unroll
    for (int i = 0; i < 4; ++i) {
      int idx = tid + NTHREADS * i;      // b*256 + tl*64 + n
      int b  = idx >> 8;
      int tl = (idx >> 6) & 3;
      int n  = idx & 63;
      const float* attn = Sc + (b * TT + tl) * H_SZ;
      const float* vcol = Cc + tl * H_SZ * NC + n;
      float y = 0.f;
#pragma unroll
      for (int h = 0; h < H_SZ; ++h) y += attn[h] * vcol[h * NC];
      size_t gi = (size_t)(b * T_SZ + t0 + tl) * D_DIM + dc + n;
      Y[gi] = y + x[gi];
    }
    __syncthreads();
  }
}

// ---------------------------------------------------------------------------
extern "C" void kernel_launch(void* const* d_in, const int* in_sizes, int n_in,
                              void* d_out, int out_size, void* d_ws, size_t ws_size,
                              hipStream_t stream) {
  const float* x       = (const float*)d_in[0];
  const float* history = (const float*)d_in[1];
  const float* Wq      = (const float*)d_in[2];
  const float* bq      = (const float*)d_in[3];
  const float* Wk      = (const float*)d_in[4];
  const float* bk      = (const float*)d_in[5];
  const float* Wv      = (const float*)d_in[6];
  const float* bv      = (const float*)d_in[7];
  const float* Wo      = (const float*)d_in[8];
  const float* bo      = (const float*)d_in[9];
  const float* decay   = (const float*)d_in[10];

  float* out = (float*)d_out;
  float* Y   = (float*)d_ws;   // 32 MB scratch: attended + x
  float* Qs  = out;            // reuse d_out as Q scratch (overwritten by step 3)

  dim3 gProj(D_DIM / NC, (B_SZ * T_SZ) / ROWS);   // (16, 64)
  proj_gemm_kernel<<<gProj, NTHREADS, 0, stream>>>(x, Wq, bq, Qs);
  history_attn_kernel<<<dim3(T_SZ / TT), NTHREADS, 0, stream>>>(
      x, history, Wk, bk, Wv, bv, decay, Qs, Y);
  proj_gemm_kernel<<<gProj, NTHREADS, 0, stream>>>(Y, Wo, bo, out);
}
